// GAT_MTL_Model_54786602828052
// MI455X (gfx1250) — compile-verified, matrix operand order checked by reading, not run on hardware
//
#include <hip/hip_runtime.h>
#include <hip/hip_bf16.h>

typedef __attribute__((ext_vector_type(16))) _Float16 v16h;
typedef __attribute__((ext_vector_type(8)))  _Float16 h8;
typedef __attribute__((ext_vector_type(8)))  float    v8f;

#define NN   50000
#define EE   800000
#define EP   (EE + NN)     // edges + self loops
#define FIN  128
#define HID  32
#define HEADS 4
#define GG   64
#define CC   5

// ---------------- conversion kernels ----------------
__global__ void cvt_f32_to_f16(const float* __restrict__ src, _Float16* __restrict__ dst, int n) {
    int t = blockIdx.x * blockDim.x + threadIdx.x;
    if (t < n) dst[t] = (_Float16)src[t];
}

// BT[n*K + k] = B[k*Ncol + n]  (convert + transpose; tiny matrices, L2 resident)
__global__ void cvt_transpose_f16(const float* __restrict__ B, _Float16* __restrict__ BT, int K, int Ncol) {
    int t = blockIdx.x * blockDim.x + threadIdx.x;
    if (t >= K * Ncol) return;
    int n = t / K, k = t % K;
    BT[t] = (_Float16)B[k * Ncol + n];
}

// ---------------- WMMA GEMM: C[M, CT*16] = A[M,K] * B, B given as BT[CT*16, K] ----------------
// One wave per 16-row tile; CT column tiles accumulated in registers so the A
// fragment is loaded once per K-step and reused across all column tiles.
template <int CT>
__global__ __launch_bounds__(256) void wmma_gemm_rt(const _Float16* __restrict__ A,
                                                    const _Float16* __restrict__ BT,
                                                    float* __restrict__ C,
                                                    int M, int K) {
    int rt   = (blockIdx.x * blockDim.x + threadIdx.x) >> 5;   // row tile = wave id
    int lane = threadIdx.x & 31;
    if (rt * 16 >= M) return;

    int m  = lane & 15;   // row (A) / col (B) owned by this lane
    int hi = lane >> 4;   // K sub-octet group (ISA 16-bit A-fragment layout)

    const _Float16* Arow = A + (size_t)(rt * 16 + m) * K;

    v8f acc[CT];
    #pragma unroll
    for (int c = 0; c < CT; ++c)
        #pragma unroll
        for (int r = 0; r < 8; ++r) acc[c][r] = 0.f;

    for (int kk = 0; kk < K; kk += 32) {
        // A lane layout: elems 0-7 <- K = kk + hi*8 + 0..7, elems 8-15 <- K = kk + 16 + hi*8 + 0..7
        h8 alo = *(const h8*)(Arow + kk + hi * 8);
        h8 ahi = *(const h8*)(Arow + kk + 16 + hi * 8);
        v16h a;
        #pragma unroll
        for (int i = 0; i < 8; ++i) { a[i] = alo[i]; a[i + 8] = ahi[i]; }

        #pragma unroll
        for (int c = 0; c < CT; ++c) {
            const _Float16* Bcol = BT + (size_t)(c * 16 + m) * K;
            h8 blo = *(const h8*)(Bcol + kk + hi * 8);
            h8 bhi = *(const h8*)(Bcol + kk + 16 + hi * 8);
            v16h b;
            #pragma unroll
            for (int i = 0; i < 8; ++i) { b[i] = blo[i]; b[i + 8] = bhi[i]; }
            acc[c] = __builtin_amdgcn_wmma_f32_16x16x32_f16(false, a, false, b,
                                                            (short)0, acc[c], false, false);
        }
    }

    // C/D layout: VGPR r -> row rt*16 + hi*8 + r, col c*16 + (lane&15)
    const int Ncol = CT * 16;
    #pragma unroll
    for (int c = 0; c < CT; ++c) {
        int col = c * 16 + m;
        #pragma unroll
        for (int r = 0; r < 8; ++r)
            C[(size_t)(rt * 16 + hi * 8 + r) * Ncol + col] = acc[c][r];
    }
}

// ---------------- attention logits per node ----------------
__global__ void al_heads(const float* __restrict__ h1, const float* __restrict__ a_src,
                         const float* __restrict__ a_dst, float* __restrict__ als,
                         float* __restrict__ ald, int n_times_h) {
    int t = blockIdx.x * blockDim.x + threadIdx.x;
    if (t >= n_times_h) return;
    int n = t >> 2, h = t & 3;           // HEADS==4
    const float* hp = h1 + (size_t)n * 128 + h * 32;
    const float* as = a_src + h * 32;
    const float* ad = a_dst + h * 32;
    float s1 = 0.f, s2 = 0.f;
    #pragma unroll
    for (int d = 0; d < 32; ++d) { float v = hp[d]; s1 += v * as[d]; s2 += v * ad[d]; }
    als[t] = s1; ald[t] = s2;
}

__global__ void al_single(const float* __restrict__ h2, const float* __restrict__ a_src,
                          const float* __restrict__ a_dst, float* __restrict__ als,
                          float* __restrict__ ald, int n) {
    int t = blockIdx.x * blockDim.x + threadIdx.x;
    if (t >= n) return;
    const float* hp = h2 + (size_t)t * 32;
    float s1 = 0.f, s2 = 0.f;
    #pragma unroll
    for (int d = 0; d < 32; ++d) { float v = hp[d]; s1 += v * a_src[d]; s2 += v * a_dst[d]; }
    als[t] = s1; ald[t] = s2;
}

// ---------------- edge pass 1: p = exp(leaky(al_s[src]+al_d[dst])), z[dst] += p ----------------
__device__ __forceinline__ float leaky02(float v) { return v > 0.f ? v : 0.2f * v; }

__global__ void edge1_logits(const int* __restrict__ ei, const float* __restrict__ als,
                             const float* __restrict__ ald, float* __restrict__ p1,
                             float* __restrict__ z1) {
    int e = blockIdx.x * blockDim.x + threadIdx.x;
    if (e >= EP) return;
    int src = (e < EE) ? ei[e] : (e - EE);
    int dst = (e < EE) ? ei[EE + e] : (e - EE);
    #pragma unroll
    for (int h = 0; h < 4; ++h) {
        float p = __expf(leaky02(als[src * 4 + h] + ald[dst * 4 + h]));
        p1[(size_t)e * 4 + h] = p;
        atomicAdd(&z1[dst * 4 + h], p);
    }
}

__global__ void edge2_logits(const int* __restrict__ ei, const float* __restrict__ als,
                             const float* __restrict__ ald, float* __restrict__ p2,
                             float* __restrict__ z2) {
    int e = blockIdx.x * blockDim.x + threadIdx.x;
    if (e >= EP) return;
    int src = (e < EE) ? ei[e] : (e - EE);
    int dst = (e < EE) ? ei[EE + e] : (e - EE);
    float p = __expf(leaky02(als[src] + ald[dst]));
    p2[e] = p;
    atomicAdd(&z2[dst], p);
}

// ---------------- edge pass 2: message scatter (wave per edge, lane = feature) ----------------
__global__ __launch_bounds__(256) void edge1_scatter(const int* __restrict__ ei,
                                                     const float* __restrict__ p1,
                                                     const float* __restrict__ h1,
                                                     float* __restrict__ acc1) {
    int e    = (blockIdx.x * blockDim.x + threadIdx.x) >> 5;
    int lane = threadIdx.x & 31;
    if (e >= EP) return;
    int src = (e < EE) ? ei[e] : (e - EE);
    int dst = (e < EE) ? ei[EE + e] : (e - EE);
    const float* hs = h1 + (size_t)src * 128;
    float* ap = acc1 + (size_t)dst * 128;
    #pragma unroll
    for (int h = 0; h < 4; ++h) {
        float p = p1[(size_t)e * 4 + h];          // broadcast load
        atomicAdd(&ap[h * 32 + lane], p * hs[h * 32 + lane]);
    }
}

__global__ __launch_bounds__(256) void edge2_scatter(const int* __restrict__ ei,
                                                     const float* __restrict__ p2,
                                                     const float* __restrict__ h2,
                                                     float* __restrict__ acc2) {
    int e    = (blockIdx.x * blockDim.x + threadIdx.x) >> 5;
    int lane = threadIdx.x & 31;
    if (e >= EP) return;
    int src = (e < EE) ? ei[e] : (e - EE);
    int dst = (e < EE) ? ei[EE + e] : (e - EE);
    float p = p2[e];
    atomicAdd(&acc2[(size_t)dst * 32 + lane], p * h2[(size_t)src * 32 + lane]);
}

// ---------------- normalize + bias + relu ----------------
__global__ void finish1(const float* __restrict__ acc1, const float* __restrict__ z1,
                        const float* __restrict__ b1, _Float16* __restrict__ hrelu) {
    int t = blockIdx.x * blockDim.x + threadIdx.x;
    if (t >= NN * 128) return;
    int n = t >> 7, d = t & 127;
    float v = acc1[t] / z1[n * 4 + (d >> 5)] + b1[d];
    hrelu[t] = (_Float16)fmaxf(v, 0.f);
}

__global__ void finish2_pool(const float* __restrict__ acc2, const float* __restrict__ z2,
                             const float* __restrict__ b2, const int* __restrict__ batch,
                             float* __restrict__ pooled, float* __restrict__ cnt) {
    int t = blockIdx.x * blockDim.x + threadIdx.x;
    if (t >= NN * 32) return;
    int n = t >> 5, d = t & 31;
    float o = fmaxf(acc2[t] / z2[n] + b2[d], 0.f);
    int g = batch[n];
    atomicAdd(&pooled[g * 32 + d], o);
    if (d == 0) atomicAdd(&cnt[g], 1.0f);
}

// ---------------- MLP heads: one thread per graph ----------------
__global__ void heads_kernel(const float* __restrict__ pooled, const float* __restrict__ cnt,
                             const float* __restrict__ dW1, const float* __restrict__ db1,
                             const float* __restrict__ dW2, const float* __restrict__ db2,
                             const float* __restrict__ sW1, const float* __restrict__ sb1,
                             const float* __restrict__ sW2, const float* __restrict__ sb2,
                             float* __restrict__ out) {
    int g = threadIdx.x;
    if (g >= GG) return;
    float inv = 1.f / fmaxf(cnt[g], 1.f);
    float pv[32];
    #pragma unroll
    for (int d = 0; d < 32; ++d) pv[d] = pooled[g * 32 + d] * inv;

    float t1[16];
    // depression head
    for (int j = 0; j < 16; ++j) {
        float s = db1[j];
        for (int d = 0; d < 32; ++d) s += pv[d] * dW1[d * 16 + j];
        t1[j] = fmaxf(s, 0.f);
    }
    float o = db2[0];
    for (int j = 0; j < 16; ++j) o += t1[j] * dW2[j];
    out[g] = 1.f / (1.f + __expf(-o));

    // severity head
    for (int j = 0; j < 16; ++j) {
        float s = sb1[j];
        for (int d = 0; d < 32; ++d) s += pv[d] * sW1[d * 16 + j];
        t1[j] = fmaxf(s, 0.f);
    }
    float sv[5], mx = -1e30f;
    for (int c = 0; c < 5; ++c) {
        float s = sb2[c];
        for (int j = 0; j < 16; ++j) s += t1[j] * sW2[j * 5 + c];
        sv[c] = s; mx = fmaxf(mx, s);
    }
    float zs = 0.f;
    for (int c = 0; c < 5; ++c) { sv[c] = __expf(sv[c] - mx); zs += sv[c]; }
    for (int c = 0; c < 5; ++c) out[GG + g * 5 + c] = sv[c] / zs;
}

// ---------------- host launcher ----------------
static inline int cdiv(long long a, int b) { return (int)((a + b - 1) / b); }

extern "C" void kernel_launch(void* const* d_in, const int* in_sizes, int n_in,
                              void* d_out, int out_size, void* d_ws, size_t ws_size,
                              hipStream_t stream) {
    const float* x      = (const float*)d_in[0];
    const int*   ei     = (const int*)d_in[1];
    const int*   batch  = (const int*)d_in[2];
    const float* W1     = (const float*)d_in[3];
    const float* a_src1 = (const float*)d_in[4];
    const float* a_dst1 = (const float*)d_in[5];
    const float* b1     = (const float*)d_in[6];
    const float* W2     = (const float*)d_in[7];
    const float* a_src2 = (const float*)d_in[8];
    const float* a_dst2 = (const float*)d_in[9];
    const float* b2     = (const float*)d_in[10];
    const float* dW1    = (const float*)d_in[11];
    const float* db1    = (const float*)d_in[12];
    const float* dW2    = (const float*)d_in[13];
    const float* db2    = (const float*)d_in[14];
    const float* sW1    = (const float*)d_in[15];
    const float* sb1    = (const float*)d_in[16];
    const float* sW2    = (const float*)d_in[17];
    const float* sb2    = (const float*)d_in[18];
    float* out = (float*)d_out;

    // workspace carve-out (256B aligned)
    size_t off = 0;
    auto alloc = [&](size_t bytes) -> char* {
        char* p = (char*)d_ws + off;
        off = (off + bytes + 255) & ~(size_t)255;
        return p;
    };
    _Float16* x16   = (_Float16*)alloc((size_t)NN * 128 * 2);
    _Float16* W1T   = (_Float16*)alloc(128 * 128 * 2);
    _Float16* W2T   = (_Float16*)alloc(32 * 128 * 2);
    float* h1       = (float*)alloc((size_t)NN * 128 * 4);
    float* als1     = (float*)alloc((size_t)NN * 4 * 4);
    float* ald1     = (float*)alloc((size_t)NN * 4 * 4);
    float* p1       = (float*)alloc((size_t)EP * 4 * 4);
    float* z1       = (float*)alloc((size_t)NN * 4 * 4);
    float* acc1     = (float*)alloc((size_t)NN * 128 * 4);
    _Float16* hrelu = (_Float16*)alloc((size_t)NN * 128 * 2);
    float* h2       = (float*)alloc((size_t)NN * 32 * 4);
    float* als2     = (float*)alloc((size_t)NN * 4);
    float* ald2     = (float*)alloc((size_t)NN * 4);
    float* p2       = (float*)alloc((size_t)EP * 4);
    float* z2       = (float*)alloc((size_t)NN * 4);
    float* acc2     = (float*)alloc((size_t)NN * 32 * 4);
    float* pooled   = (float*)alloc(GG * 32 * 4);
    float* cnt      = (float*)alloc(GG * 4);

    // zero accumulators every call (ws is not re-poisoned between replays)
    hipMemsetAsync(z1, 0, (size_t)NN * 4 * 4, stream);
    hipMemsetAsync(acc1, 0, (size_t)NN * 128 * 4, stream);
    hipMemsetAsync(z2, 0, (size_t)NN * 4, stream);
    hipMemsetAsync(acc2, 0, (size_t)NN * 32 * 4, stream);
    hipMemsetAsync(pooled, 0, GG * 32 * 4, stream);
    hipMemsetAsync(cnt, 0, GG * 4, stream);

    const int B = 256;

    // conversions
    cvt_f32_to_f16<<<cdiv((long long)NN * 128, B), B, 0, stream>>>(x, x16, NN * 128);
    cvt_transpose_f16<<<cdiv(128 * 128, B), B, 0, stream>>>(W1, W1T, 128, 128);
    cvt_transpose_f16<<<cdiv(32 * 128, B), B, 0, stream>>>(W2, W2T, 128, 32);

    // ---- GAT layer 1 ----
    {   // one wave per 16-row tile, 8 column tiles in registers
        int waves = NN / 16;   // 3125
        wmma_gemm_rt<8><<<cdiv(waves, 8), B, 0, stream>>>(x16, W1T, h1, NN, 128);
    }
    al_heads<<<cdiv(NN * 4, B), B, 0, stream>>>(h1, a_src1, a_dst1, als1, ald1, NN * 4);
    edge1_logits<<<cdiv(EP, B), B, 0, stream>>>(ei, als1, ald1, p1, z1);
    edge1_scatter<<<cdiv((long long)EP * 32, B), B, 0, stream>>>(ei, p1, h1, acc1);
    finish1<<<cdiv((long long)NN * 128, B), B, 0, stream>>>(acc1, z1, b1, hrelu);

    // ---- GAT layer 2 ----
    {   int waves = NN / 16;   // 3125, 2 column tiles
        wmma_gemm_rt<2><<<cdiv(waves, 8), B, 0, stream>>>(hrelu, W2T, h2, NN, 128);
    }
    al_single<<<cdiv(NN, B), B, 0, stream>>>(h2, a_src2, a_dst2, als2, ald2, NN);
    edge2_logits<<<cdiv(EP, B), B, 0, stream>>>(ei, als2, ald2, p2, z2);
    edge2_scatter<<<cdiv((long long)EP * 32, B), B, 0, stream>>>(ei, p2, h2, acc2);
    finish2_pool<<<cdiv((long long)NN * 32, B), B, 0, stream>>>(acc2, z2, b2, batch, pooled, cnt);

    // ---- heads ----
    heads_kernel<<<1, 64, 0, stream>>>(pooled, cnt, dW1, db1, dW2, db2, sW1, sb1, sW2, sb2, out);
}
